// SNN_2044404432896
// MI455X (gfx1250) — matrix operand order repaired and checked by reading, not compiled, not run on hardware
//
#include <hip/hip_runtime.h>

// ---------------------------------------------------------------------------
// SNN forward: xin = x @ w_in^T  (huge memory-bound GEMM, WMMA f16->f32),
// then a tiny sequential LIF/LI recurrence in a single persistent workgroup.
// M-blocked 2x: each B (w_in) fragment feeds 2 WMMAs (L2 traffic 9.2GB->4.6GB).
// 128-thread blocks (4 waves); amdgpu_waves_per_eu(1,1) pins the occupancy
// target so the register allocator gives the wave the whole VGPR budget --
// all 14 accumulators stay register-resident (no scratch in the hot loop).
// sched_barriers cap the B-load staging window at 2 sets.
// ---------------------------------------------------------------------------

typedef __attribute__((ext_vector_type(16))) _Float16 v16h;
typedef __attribute__((ext_vector_type(8)))  float    v8f;

#if defined(__has_builtin)
#if __has_builtin(__builtin_amdgcn_sched_barrier)
#define SCHED_FENCE() __builtin_amdgcn_sched_barrier(0)
#endif
#endif
#ifndef SCHED_FENCE
#define SCHED_FENCE()
#endif

#define T_DIM   50
#define B_DIM   16
#define F_DIM   460800      // 2*640*360
#define H_DIM   100
#define O_DIM   6
#define NPAD    112         // 7 tiles of 16
#define NTILES  7
#define MTILES  2           // 2 m-tiles (= 2 t values, 32 rows) per block
#define MGROUPS 25          // 50 / 2, exact
#define GWAVES  4           // waves per GEMM block (128 threads)
#define KSPLITS 60
#define KCHUNK  (F_DIM / KSPLITS)          // 7680, multiple of 32
#define KSTEPS  (KCHUNK / (GWAVES * 32))   // 60 k-steps of 32 per wave

__device__ __forceinline__ float4 ldg4(const float* p) {
  return *reinterpret_cast<const float4*>(p);
}

__device__ __forceinline__ void pack8(v16h& d, int base, const float4& a, const float4& b) {
  d[base + 0] = (_Float16)a.x;
  d[base + 1] = (_Float16)a.y;
  d[base + 2] = (_Float16)a.z;
  d[base + 3] = (_Float16)a.w;
  d[base + 4] = (_Float16)b.x;
  d[base + 5] = (_Float16)b.y;
  d[base + 6] = (_Float16)b.z;
  d[base + 7] = (_Float16)b.w;
}

// ---------------------------------------------------------------------------
// Kernel 0: zero the split-K accumulator (d_ws is poisoned by harness).
// ---------------------------------------------------------------------------
__global__ void snn_zero_ws(float* __restrict__ p, int n) {
  int i = blockIdx.x * blockDim.x + threadIdx.x;
  if (i < n) p[i] = 0.0f;
}

// ---------------------------------------------------------------------------
// Kernel 1: xin[t,b,h] += sum_k x[t,b,k] * w_in[h,k] over one K chunk.
// grid = (25 m-groups, 60 k-splits), block = 128 (4 waves).
// Each block covers 2 t's (32 rows) and all 7 N tiles; B frags reused 2x.
// ---------------------------------------------------------------------------
__global__ __launch_bounds__(128, 1)
__attribute__((amdgpu_waves_per_eu(1, 1)))
void snn_in_gemm(
    const float* __restrict__ x,      // [800, F]
    const float* __restrict__ w_in,   // [100, F]
    float* __restrict__ xin) {        // [800, 112] accumulator
  const int g    = blockIdx.x;        // m-group: t = 2g, 2g+1
  const int kc   = blockIdx.y;
  const int tid  = threadIdx.x;
  const int wave = tid >> 5;
  const int lane = tid & 31;
  const int m    = lane & 15;   // row within 16x16 tile
  const int hh   = lane >> 4;   // half-wave select (WMMA fragment layout)

  // 32-bit element offsets off the uniform bases (saddr addressing).
  // A fragment (16x32 f16): lane half hh -> K runs [8hh,8hh+7], [16+8hh,16+8hh+7]
  const unsigned aoff0 = (unsigned)((g * 2 + 0) * 16 + m) * (unsigned)F_DIM + 8u * hh;
  const unsigned aoff1 = (unsigned)((g * 2 + 1) * 16 + m) * (unsigned)F_DIM + 8u * hh;

  // B fragment (32x16 f16): lane half hh -> K runs [8hh,8hh+7], [16+8hh,16+8hh+7]
  bool nok[NTILES];
  unsigned boff[NTILES];
  #pragma unroll
  for (int j = 0; j < NTILES; ++j) {
    int n  = j * 16 + m;               // w_in row (= output column)
    nok[j] = (n < H_DIM);
    boff[j] = (unsigned)(nok[j] ? n : 0) * (unsigned)F_DIM + 8u * hh;
  }

  v8f acc[MTILES][NTILES];
  #pragma unroll
  for (int mt = 0; mt < MTILES; ++mt)
    #pragma unroll
    for (int j = 0; j < NTILES; ++j)
      acc[mt][j] = (v8f){0.f, 0.f, 0.f, 0.f, 0.f, 0.f, 0.f, 0.f};

  const unsigned kbase = (unsigned)kc * KCHUNK + (unsigned)wave * 32u;

  #pragma unroll 1
  for (int s = 0; s < KSTEPS; ++s) {
    const unsigned koff = kbase + (unsigned)s * (GWAVES * 32u);

    // ---- load + convert the 2 A fragments ----
    v16h af[MTILES];
    {
      const float* ap = x + (aoff0 + koff);
      float4 a0 = ldg4(ap), a1 = ldg4(ap + 4), a2 = ldg4(ap + 16), a3 = ldg4(ap + 20);
      pack8(af[0], 0, a0, a1);
      pack8(af[0], 8, a2, a3);
      __builtin_prefetch(ap + GWAVES * 32, 0, 3);  // global_prefetch_b8 (speculative)
    }
    {
      const float* ap = x + (aoff1 + koff);
      float4 a0 = ldg4(ap), a1 = ldg4(ap + 4), a2 = ldg4(ap + 16), a3 = ldg4(ap + 20);
      pack8(af[1], 0, a0, a1);
      pack8(af[1], 8, a2, a3);
      __builtin_prefetch(ap + GWAVES * 32, 0, 3);
    }

    // ---- for each n-tile: one B fragment feeds 2 WMMAs ----
    #pragma unroll
    for (int j = 0; j < NTILES; ++j) {
      const float* bp = w_in + (boff[j] + koff);
      float4 b0, b1, b2, b3;
      if (nok[j]) {
        b0 = ldg4(bp);
        b1 = ldg4(bp + 4);
        b2 = ldg4(bp + 16);
        b3 = ldg4(bp + 20);
      } else {
        b0 = b1 = b2 = b3 = make_float4(0.f, 0.f, 0.f, 0.f);
      }
      v16h bf;
      pack8(bf, 0, b0, b1);
      pack8(bf, 8, b2, b3);

      #pragma unroll
      for (int mt = 0; mt < MTILES; ++mt) {
        acc[mt][j] = __builtin_amdgcn_wmma_f32_16x16x32_f16(
            /*neg_a=*/false, af[mt], /*neg_b=*/false, bf,
            /*c_mod=*/(short)0, acc[mt][j], /*reuse_a=*/false, /*reuse_b=*/false);
      }

      // Cap the load-hoisting window: at most 2 B staging sets live at once.
      if ((j & 1) == 1) SCHED_FENCE();
    }
  }

  // Cross-wave reduction in LDS (ds_add_f32), then split-K global atomics.
  // One 7KB buffer reused across the 2 m-tiles.
  __shared__ float red[NTILES * 256];
  #pragma unroll 1
  for (int mt = 0; mt < MTILES; ++mt) {
    const int t = g * MTILES + mt;

    for (int idx = tid; idx < NTILES * 256; idx += 128) red[idx] = 0.0f;
    __syncthreads();

    #pragma unroll
    for (int j = 0; j < NTILES; ++j) {
      #pragma unroll
      for (int r = 0; r < 8; ++r) {
        // C layout: VGPR r holds row (r + 8*hh), column m
        int mrow = r + 8 * hh;
        atomicAdd(&red[j * 256 + mrow * 16 + m], acc[mt][j][r]);
      }
    }
    __syncthreads();

    for (int idx = tid; idx < NTILES * 256; idx += 128) {
      int j = idx >> 8;
      int e = idx & 255;
      int mrow = e >> 4;
      int col  = e & 15;
      atomicAdd(&xin[((size_t)t * 16 + mrow) * NPAD + j * 16 + col], red[idx]);
    }
    __syncthreads();   // red reads done before next m-tile zeroes it
  }
}

// ---------------------------------------------------------------------------
// Kernel 2: sequential LIF/LI recurrence. One block, 256 threads.
// State: z in LDS (shared for the 2 small GEMMs); v,i,vo,io in registers.
// ---------------------------------------------------------------------------
__global__ __launch_bounds__(256) void snn_recur(
    const float* __restrict__ xin,    // [800, 112]
    const float* __restrict__ w_rec,  // [100, 100]
    const float* __restrict__ w_out,  // [6, 100]
    float* __restrict__ out) {        // [50, 16, 6]
  __shared__ float s_wrec[H_DIM * H_DIM];   // 40 KB
  __shared__ float s_wout[O_DIM * H_DIM];
  __shared__ float s_z[B_DIM * H_DIM];      // spikes (shared across waves)

  const int tid = threadIdx.x;

  for (int i = tid; i < H_DIM * H_DIM; i += 256) s_wrec[i] = w_rec[i];
  for (int i = tid; i < O_DIM * H_DIM; i += 256) s_wout[i] = w_out[i];

  // Each thread owns bh = tid, tid+256, ... (<= 7 entries of the 1600).
  float v_r[7], i_r[7], rec_r[7];
  {
    int c = 0;
    for (int bh = tid; bh < B_DIM * H_DIM; bh += 256, ++c) {
      v_r[c] = 0.0f;
      i_r[c] = 0.0f;
      s_z[bh] = 0.0f;
    }
  }
  float vo = 0.0f, io = 0.0f;
  const int b_o = tid / O_DIM;   // valid when tid < 96
  const int o_o = tid % O_DIM;
  __syncthreads();

  for (int t = 0; t < T_DIM; ++t) {
    // 1) recurrent drive from OLD z:  rec[b,h] = sum_h2 z[b,h2] * w_rec[h,h2]
    {
      int c = 0;
      for (int bh = tid; bh < B_DIM * H_DIM; bh += 256, ++c) {
        int b = bh / H_DIM;
        int h = bh % H_DIM;
        const float* zrow = &s_z[b * H_DIM];
        const float* wrow = &s_wrec[h * H_DIM];
        float a = 0.0f;
        for (int k = 0; k < H_DIM; ++k) a = fmaf(zrow[k], wrow[k], a);
        rec_r[c] = a;
      }
    }
    __syncthreads();   // all reads of old z done before overwrite

    // 2) LIF update (Euler): v_dec uses pre-decay current, reset to 0 on spike
    {
      int c = 0;
      for (int bh = tid; bh < B_DIM * H_DIM; bh += 256, ++c) {
        int b = bh / H_DIM;
        int h = bh % H_DIM;
        float vdec = v_r[c] + 0.1f * (i_r[c] - v_r[c]);    // DT*TAU_MEM_INV = 0.1
        float zn   = (vdec - 0.3f) > 0.0f ? 1.0f : 0.0f;   // V_TH = 0.3
        v_r[c] = (1.0f - zn) * vdec;
        i_r[c] = 0.9f * i_r[c]                              // 1 - DT*TAU_SYN_INV
               + xin[((size_t)t * B_DIM + b) * NPAD + h]
               + rec_r[c];
        s_z[bh] = zn;                                       // publish new z
      }
    }
    __syncthreads();   // new z visible for readout

    // 3) readout + LI cell; output is vo_new
    if (tid < B_DIM * O_DIM) {
      const float* zrow = &s_z[b_o * H_DIM];
      const float* wrow = &s_wout[o_o * H_DIM];
      float y = 0.0f;
      for (int k = 0; k < H_DIM; ++k) y = fmaf(zrow[k], wrow[k], y);
      float von = vo + 0.1f * (io - vo);                    // DT*LI_TAU_MEM_INV
      out[((size_t)t * B_DIM + b_o) * O_DIM + o_o] = von;
      io = 0.8f * io + y;                                   // 1 - DT*LI_TAU_SYN_INV
      vo = von;
    }
    __syncthreads();
  }
}

// ---------------------------------------------------------------------------
extern "C" void kernel_launch(void* const* d_in, const int* in_sizes, int n_in,
                              void* d_out, int out_size, void* d_ws, size_t ws_size,
                              hipStream_t stream) {
  const float* x     = (const float*)d_in[0];  // [50,16,2,640,360] fp32
  const float* w_in  = (const float*)d_in[1];  // [100, 460800]
  const float* w_rec = (const float*)d_in[2];  // [100, 100]
  const float* w_out = (const float*)d_in[3];  // [6, 100]
  float* out = (float*)d_out;                  // [50,16,6]
  float* xin = (float*)d_ws;                   // [800, 112] accumulator

  const int xin_elems = T_DIM * B_DIM * NPAD;  // 89600
  snn_zero_ws<<<dim3((xin_elems + 255) / 256), dim3(256), 0, stream>>>(xin, xin_elems);

  snn_in_gemm<<<dim3(MGROUPS, KSPLITS), dim3(128), 0, stream>>>(x, w_in, xin);

  snn_recur<<<dim3(1), dim3(256), 0, stream>>>(xin, w_rec, w_out, out);
}
